// CaptionDecoder_20701742367076
// MI455X (gfx1250) — matrix operand order repaired and checked by reading, not compiled, not run on hardware
//
#include <hip/hip_runtime.h>
#include <hip/hip_bf16.h>
#include <stdint.h>

// ---- problem dims ----
#define VV    10000
#define EE    300
#define AA    512
#define ENC   512
#define DD    512
#define BB    128
#define NN    196
#define LL    25
#define TT    24          // L-1 decode steps
#define G4D   2048        // 4*D
#define KCAT  1344        // 512 (h) + 300 (emb) + 512 (ctx) + 20 pad  (multiple of 32, ktiles=42 even)

typedef float  v8f   __attribute__((ext_vector_type(8)));
typedef __bf16 v16bf __attribute__((ext_vector_type(16)));

union Frag { uint4 u[2]; v16bf v; };

__device__ __forceinline__ unsigned short f2bf(float x) {
    union { float f; unsigned u; } v; v.f = x;
    unsigned r = v.u + 0x7FFFu + ((v.u >> 16) & 1u);   // round-nearest-even
    return (unsigned short)(r >> 16);
}

// ---------------- preprocessing kernels ----------------

__global__ void k_cvt_bf16(const float* __restrict__ src, unsigned short* __restrict__ dst, int n) {
    int i = blockIdx.x * blockDim.x + threadIdx.x;
    if (i < n) dst[i] = f2bf(src[i]);
}

// Swizzle a row-major f32 [K,N] weight matrix into the contiguous-k WMMA B-fragment bf16
// layout: dst[((nt*ktiles + kt)*32 + lane)*16 + hh], element = B[kt*32 + 16*(lane>>4) + hh, nt*16 + (lane&15)]
__global__ void k_swizzleB(const float* __restrict__ src, unsigned short* __restrict__ dst, int K, int N) {
    int id = blockIdx.x * blockDim.x + threadIdx.x;
    int ktiles = K >> 5, ntiles = N >> 4;
    int total = ktiles * ntiles * 32;
    if (id >= total) return;
    int lane = id & 31;
    int tile = id >> 5;
    int kt = tile % ktiles;
    int nt = tile / ktiles;
    int n  = nt * 16 + (lane & 15);
    int kb = kt * 32 + 16 * (lane >> 4);
    unsigned short* o = dst + (size_t)id * 16;
#pragma unroll
    for (int hh = 0; hh < 16; ++hh)
        o[hh] = f2bf(src[(size_t)(kb + hh) * N + n]);
}

// Build padded concatenated gate weights Wcat[KCAT, 4D] (f32):
// rows 0..511 = W_hh, 512..811 = W_ih(emb rows), 812..1323 = W_ih(ctx rows), 1324..1343 = 0
__global__ void k_build_wcat(const float* __restrict__ W_ih, const float* __restrict__ W_hh,
                             float* __restrict__ dst) {
    int i = blockIdx.x * blockDim.x + threadIdx.x;
    if (i >= KCAT * G4D) return;
    int k = i / G4D, n = i - k * G4D;
    float v = 0.f;
    if (k < DD)                 v = W_hh[(size_t)k * G4D + n];
    else if (k < DD + EE + ENC) v = W_ih[(size_t)(k - DD) * G4D + n];
    dst[i] = v;
}

__global__ void k_bias_cat(const float* __restrict__ b_ih, const float* __restrict__ b_hh,
                           float* __restrict__ dst) {
    int i = blockIdx.x * blockDim.x + threadIdx.x;
    if (i < G4D) dst[i] = b_ih[i] + b_hh[i];
}

__global__ void k_gather_emb(const float* __restrict__ embedding, const int* __restrict__ captions,
                             unsigned short* __restrict__ emb_bf) {
    int i = blockIdx.x * blockDim.x + threadIdx.x;
    if (i >= BB * LL * EE) return;
    int e = i % EE;
    int l = (i / EE) % LL;
    int b = i / (EE * LL);
    int tok = captions[b * LL + l];
    emb_bf[i] = f2bf(embedding[(size_t)tok * EE + e]);
}

__global__ void k_mean(const float* __restrict__ f, float* __restrict__ mean) {
    int b = blockIdx.x;
    for (int d = threadIdx.x; d < ENC; d += blockDim.x) {
        float s = 0.f;
        const float* p = f + (size_t)b * NN * ENC + d;
        for (int n = 0; n < NN; ++n) s += p[(size_t)n * ENC];
        mean[b * ENC + d] = s * (1.0f / NN);
    }
}

__global__ void k_init_hc(const float* __restrict__ mean,
                          const float* __restrict__ Wi_h, const float* __restrict__ bi_h,
                          const float* __restrict__ Wi_c, const float* __restrict__ bi_c,
                          float* __restrict__ h, float* __restrict__ c,
                          unsigned short* __restrict__ xh) {
    int id = blockIdx.x * blockDim.x + threadIdx.x;       // BB*DD = 65536
    if (id >= BB * DD) return;
    int b = id >> 9, d = id & 511;
    const float* mr = mean + b * ENC;
    float sh = bi_h[d], sc = bi_c[d];
    for (int k = 0; k < ENC; ++k) {
        float m = mr[k];
        sh += m * Wi_h[(size_t)k * DD + d];
        sc += m * Wi_c[(size_t)k * DD + d];
    }
    h[id] = sh; c[id] = sc;
    xh[(size_t)b * KCAT + d] = f2bf(sh);                  // h slice = cols [0,512)
}

// ---------------- WMMA GEMM ----------------
// C[M,N](f32,+bias) = A[M,K](bf16 row-major) x Bsw (pre-swizzled bf16).
// Wave tile 16x64 (4 accumulators), block tile 64x128 (8 waves).
// Requirements: ktiles = K/32 even; M multiple of 16 per active wave; N multiple of 16.
// Branchless K-loop (OOB N-subtiles clamp their B tile; only stores are guarded) so
// EXEC stays all-ones across every v_wmma. Ping-pong fragments software-pipeline the loads.

__device__ __forceinline__ void loadA(Frag& f, const unsigned short* p) {
    f.u[0] = *(const uint4*)(p);
    f.u[1] = *(const uint4*)(p + 8);
}
__device__ __forceinline__ void loadB(Frag& f, const unsigned short* p) {
    f.u[0] = *(const uint4*)(p);
    f.u[1] = *(const uint4*)(p + 8);
}

__global__ __launch_bounds__(256)
void k_gemm_bf16(const unsigned short* __restrict__ A,
                 const unsigned short* __restrict__ Bsw,
                 const float* __restrict__ bias,
                 float* __restrict__ C,
                 int M, int N, int K, int lda, int ldc) {
    const int lane = threadIdx.x & 31;
    const int wave = __builtin_amdgcn_readfirstlane(threadIdx.x >> 5);  // scalarize
    const int wm = wave & 3, wn = wave >> 2;              // 4 (M) x 2 (N) waves
    const int ktiles = K >> 5;
    const int m0 = blockIdx.y * 64 + wm * 16;
    const int n0 = blockIdx.x * 128 + wn * 64;
    if (m0 + 16 > M) return;                              // scalar branch
    int jmax = (N - n0) >> 4; if (jmax > 4) jmax = 4;     // #valid 16-col subtiles (may be <=0)
    const int g  = lane >> 4;
    const int ml = lane & 15;
    const unsigned short* arow = A + (size_t)(m0 + ml) * lda + 16 * g;

    // Per-subtile B base pointers; OOB subtiles clamp to the last valid tile (loads stay
    // in-bounds, results discarded at store time). Stride per K-step: 32 lanes * 16 halfs.
    const int nt_last = (N >> 4) - 1;
    const unsigned short* bbase[4];
#pragma unroll
    for (int j = 0; j < 4; ++j) {
        int nt = (n0 >> 4) + j;
        if (nt > nt_last) nt = nt_last;
        bbase[j] = Bsw + ((size_t)nt * ktiles * 32 + lane) * 16;
    }

    v8f acc[4];
#pragma unroll
    for (int j = 0; j < 4; ++j)
#pragma unroll
        for (int r = 0; r < 8; ++r) acc[j][r] = 0.f;

    // prime pipeline (kt = 0)
    Frag a0, a1, b0[4], b1[4];
    loadA(a0, arow);
#pragma unroll
    for (int j = 0; j < 4; ++j) loadB(b0[j], bbase[j]);

    for (int kt = 0; kt < ktiles; kt += 2) {              // ktiles even
        // issue loads for kt+1 while kt's data is multiplied
        loadA(a1, arow + (size_t)(kt + 1) * 32);
#pragma unroll
        for (int j = 0; j < 4; ++j) loadB(b1[j], bbase[j] + (size_t)(kt + 1) * 512);
        __builtin_prefetch(arow + (size_t)(kt + 2) * 32, 0, 3);
#pragma unroll
        for (int j = 0; j < 4; ++j)
            acc[j] = __builtin_amdgcn_wmma_f32_16x16x32_bf16(
                         false, a0.v, false, b0[j].v, (short)0, acc[j], false, false);
        if (kt + 2 < ktiles) {                            // scalar branch
            loadA(a0, arow + (size_t)(kt + 2) * 32);
#pragma unroll
            for (int j = 0; j < 4; ++j) loadB(b0[j], bbase[j] + (size_t)(kt + 2) * 512);
        }
#pragma unroll
        for (int j = 0; j < 4; ++j)
            acc[j] = __builtin_amdgcn_wmma_f32_16x16x32_bf16(
                         false, a1.v, false, b1[j].v, (short)0, acc[j], false, false);
    }

#pragma unroll
    for (int j = 0; j < 4; ++j) {
        if (j < jmax) {                                   // scalar guard, stores only
            int nc = n0 + 16 * j + ml;
            float bv = bias ? bias[nc] : 0.f;
#pragma unroll
            for (int r = 0; r < 8; ++r) {
                int row = m0 + r + 8 * g;
                C[(size_t)row * ldc + nc] = acc[j][r] + bv;
            }
        }
    }
}

// ---------------- per-step attention (one block per batch element) ----------------

__global__ __launch_bounds__(256)
void k_attention(const float* __restrict__ featproj, const float* __restrict__ features,
                 const float* __restrict__ h,
                 const float* __restrict__ W_dec, const float* __restrict__ b_dec,
                 const float* __restrict__ w_att, const float* __restrict__ b_att,
                 const unsigned short* __restrict__ emb_bf,
                 float* __restrict__ alphas_out, unsigned short* __restrict__ xh, int t) {
    __shared__ float sh_h[DD];
    __shared__ float sh_dp[AA];
    __shared__ float sh_sc[NN];
    const int b = blockIdx.x;
    const int tid = threadIdx.x;
    const int lane = tid & 31, wave = tid >> 5;

    for (int i = tid; i < DD; i += 256) sh_h[i] = h[b * DD + i];
    __syncthreads();

    // dec_proj = h @ W_dec + b_dec
    for (int a = tid; a < AA; a += 256) {
        float s = b_dec[a];
        for (int d = 0; d < DD; ++d) s += sh_h[d] * W_dec[(size_t)d * AA + a];
        sh_dp[a] = s;
    }
    __syncthreads();

    // scores[n] = sum_a tanh(featproj[b,n,a] + dec_proj[a]) * w_att[a] + b_att
    for (int n = wave; n < NN; n += 8) {
        const float* fp = featproj + ((size_t)b * NN + n) * AA;
        float s = 0.f;
        for (int a = lane; a < AA; a += 32) s += tanhf(fp[a] + sh_dp[a]) * w_att[a];
#pragma unroll
        for (int o = 16; o; o >>= 1) s += __shfl_xor(s, o, 32);
        if (lane == 0) sh_sc[n] = s + b_att[0];
    }
    __syncthreads();

    // softmax over N (wave 0)
    if (wave == 0) {
        float m = -1e30f;
        for (int n = lane; n < NN; n += 32) m = fmaxf(m, sh_sc[n]);
#pragma unroll
        for (int o = 16; o; o >>= 1) m = fmaxf(m, __shfl_xor(m, o, 32));
        float sum = 0.f;
        for (int n = lane; n < NN; n += 32) { float e = __expf(sh_sc[n] - m); sh_sc[n] = e; sum += e; }
#pragma unroll
        for (int o = 16; o; o >>= 1) sum += __shfl_xor(sum, o, 32);
        float inv = 1.f / sum;
        for (int n = lane; n < NN; n += 32) sh_sc[n] *= inv;
    }
    __syncthreads();

    for (int n = tid; n < NN; n += 256)
        alphas_out[((size_t)b * TT + t) * NN + n] = sh_sc[n];

    // context[d] = sum_n alpha[n] * features[b,n,d]  -> bf16 into xh cols [812,1324)
    for (int d = tid; d < ENC; d += 256) {
        float s = 0.f;
        const float* fb = features + (size_t)b * NN * ENC + d;
        for (int n = 0; n < NN; ++n) s += sh_sc[n] * fb[(size_t)n * ENC];
        xh[(size_t)b * KCAT + DD + EE + d] = f2bf(s);
    }
    // emb row -> cols [512,812); zero pad -> cols [1324,1344)
    for (int e = tid; e < EE; e += 256)
        xh[(size_t)b * KCAT + DD + e] = emb_bf[((size_t)b * LL + t) * EE + e];
    for (int p = tid; p < KCAT - (DD + EE + ENC); p += 256)
        xh[(size_t)b * KCAT + DD + EE + ENC + p] = 0;
}

// ---------------- LSTM elementwise ----------------

__global__ void k_lstm_elem(const float* __restrict__ gates, float* __restrict__ c,
                            float* __restrict__ h, unsigned short* __restrict__ xh) {
    int id = blockIdx.x * blockDim.x + threadIdx.x;       // BB*DD
    if (id >= BB * DD) return;
    int b = id >> 9, d = id & 511;
    const float* gr = gates + (size_t)b * G4D;
    float ig = gr[d], fg = gr[DD + d], gg = gr[2 * DD + d], og = gr[3 * DD + d];
    float si = 1.f / (1.f + __expf(-ig));
    float sf = 1.f / (1.f + __expf(-fg));
    float so = 1.f / (1.f + __expf(-og));
    float cn = sf * c[id] + si * tanhf(gg);
    float hn = so * tanhf(cn);
    c[id] = cn; h[id] = hn;
    xh[(size_t)b * KCAT + d] = f2bf(hn);                  // feed next step's GEMMs
}

// ---------------- host launcher ----------------

extern "C" void kernel_launch(void* const* d_in, const int* in_sizes, int n_in,
                              void* d_out, int out_size, void* d_ws, size_t ws_size,
                              hipStream_t stream) {
    const float* features  = (const float*)d_in[0];
    const int*   captions  = (const int*)  d_in[1];
    const float* embedding = (const float*)d_in[2];
    const float* W_enc = (const float*)d_in[3];
    const float* b_enc = (const float*)d_in[4];
    const float* W_dec = (const float*)d_in[5];
    const float* b_dec = (const float*)d_in[6];
    const float* w_att = (const float*)d_in[7];
    const float* b_att = (const float*)d_in[8];
    const float* Wi_h  = (const float*)d_in[9];
    const float* bi_h  = (const float*)d_in[10];
    const float* Wi_c  = (const float*)d_in[11];
    const float* bi_c  = (const float*)d_in[12];
    const float* W_ih  = (const float*)d_in[13];
    const float* b_ih  = (const float*)d_in[14];
    const float* W_hh  = (const float*)d_in[15];
    const float* b_hh  = (const float*)d_in[16];
    const float* W_out = (const float*)d_in[17];
    const float* b_out = (const float*)d_in[18];

    float* preds  = (float*)d_out;                               // [B,T,V]
    float* alphas = preds + (size_t)BB * TT * VV;                // [B,T,N]

    // workspace carve-up (256B aligned)
    char* ws = (char*)d_ws;
    size_t off = 0;
    auto carve = [&](size_t bytes) { void* p = ws + off; off += (bytes + 255) & ~(size_t)255; return p; };
    unsigned short* feat_bf  = (unsigned short*)carve((size_t)BB * NN * ENC * 2);
    float*          featproj = (float*)         carve((size_t)BB * NN * AA * 4);
    unsigned short* Wenc_sw  = (unsigned short*)carve((size_t)ENC * AA * 2);
    float*          Wcat     = (float*)         carve((size_t)KCAT * G4D * 4);
    unsigned short* Wcat_sw  = (unsigned short*)carve((size_t)KCAT * G4D * 2);
    unsigned short* Wout_sw  = (unsigned short*)carve((size_t)DD * VV * 2);
    unsigned short* emb_bf   = (unsigned short*)carve((size_t)BB * LL * EE * 2);
    unsigned short* xh       = (unsigned short*)carve((size_t)BB * KCAT * 2);
    float*          hbuf     = (float*)         carve((size_t)BB * DD * 4);
    float*          cbuf     = (float*)         carve((size_t)BB * DD * 4);
    float*          meanb    = (float*)         carve((size_t)BB * ENC * 4);
    float*          gates    = (float*)         carve((size_t)BB * G4D * 4);
    float*          biascat  = (float*)         carve((size_t)G4D * 4);
    (void)ws_size; (void)n_in; (void)in_sizes; (void)out_size;

    // ---- one-time preprocessing ----
    {
        int n = BB * NN * ENC;
        k_cvt_bf16<<<(n + 255) / 256, 256, 0, stream>>>(features, feat_bf, n);
    }
    {
        int n = (ENC / 32) * (AA / 16) * 32;
        k_swizzleB<<<(n + 255) / 256, 256, 0, stream>>>(W_enc, Wenc_sw, ENC, AA);
    }
    {
        int n = KCAT * G4D;
        k_build_wcat<<<(n + 255) / 256, 256, 0, stream>>>(W_ih, W_hh, Wcat);
        int m = (KCAT / 32) * (G4D / 16) * 32;
        k_swizzleB<<<(m + 255) / 256, 256, 0, stream>>>(Wcat, Wcat_sw, KCAT, G4D);
    }
    {
        int n = (DD / 32) * (VV / 16) * 32;
        k_swizzleB<<<(n + 255) / 256, 256, 0, stream>>>(W_out, Wout_sw, DD, VV);
    }
    k_bias_cat<<<(G4D + 255) / 256, 256, 0, stream>>>(b_ih, b_hh, biascat);
    {
        int n = BB * LL * EE;
        k_gather_emb<<<(n + 255) / 256, 256, 0, stream>>>(embedding, captions, emb_bf);
    }
    k_mean<<<BB, 256, 0, stream>>>(features, meanb);
    k_init_hc<<<(BB * DD + 255) / 256, 256, 0, stream>>>(meanb, Wi_h, bi_h, Wi_c, bi_c,
                                                         hbuf, cbuf, xh);
    // feat_proj = features @ W_enc + b_enc   [25088, 512], ktiles = 16
    k_gemm_bf16<<<dim3(AA / 128, (BB * NN) / 64), 256, 0, stream>>>(
        feat_bf, Wenc_sw, b_enc, featproj, BB * NN, AA, ENC, ENC, AA);

    // ---- decode scan ----
    for (int t = 0; t < TT; ++t) {
        k_attention<<<BB, 256, 0, stream>>>(featproj, features, hbuf, W_dec, b_dec,
                                            w_att, b_att, emb_bf, alphas, xh, t);
        // gates = [h|emb|ctx|0] @ Wcat + (b_ih + b_hh)   [128, 2048], ktiles = 42
        k_gemm_bf16<<<dim3(G4D / 128, BB / 64), 256, 0, stream>>>(
            xh, Wcat_sw, biascat, gates, BB, G4D, KCAT, KCAT, G4D);
        k_lstm_elem<<<(BB * DD + 255) / 256, 256, 0, stream>>>(gates, cbuf, hbuf, xh);
        // preds[:,t,:] = h_new @ W_out + b_out   [128, 10000], A = h slice of xh, ktiles = 16
        k_gemm_bf16<<<dim3((VV + 127) / 128, BB / 64), 256, 0, stream>>>(
            xh, Wout_sw, b_out, preds + (size_t)t * VV, BB, VV, DD, KCAT, (size_t)TT * VV);
    }
}
